// ConvLayer_31671088841250
// MI455X (gfx1250) — matrix operand order repaired and checked by reading, not compiled, not run on hardware
//
#include <hip/hip_runtime.h>

#define NN 10000
#define EE 160000
#define ST 4   // 16-row sub-tiles per wave (M = 64 edges/wave)

typedef __attribute__((ext_vector_type(16))) __bf16 v16bf;
typedef __attribute__((ext_vector_type(8)))  float  v8f;

union FragU { v16bf v; uint4 q[2]; unsigned u[8]; };

__device__ __forceinline__ v8f wmma_bf16(v16bf a, v16bf b, v8f c) {
  return __builtin_amdgcn_wmma_f32_16x16x32_bf16(false, a, false, b, (short)0, c, false, false);
}

__device__ __forceinline__ unsigned pk_bf16(float lo, float hi) {
  unsigned ul = __float_as_uint(lo), uh = __float_as_uint(hi);
  ul = (ul + 0x7FFFu + ((ul >> 16) & 1u)) >> 16;
  uh = (uh + 0x7FFFu + ((uh >> 16) & 1u)) & 0xFFFF0000u;
  return uh | (ul & 0xFFFFu);
}
__device__ __forceinline__ unsigned short bf16s(float f) {
  unsigned u = __float_as_uint(f);
  return (unsigned short)((u + 0x7FFFu + ((u >> 16) & 1u)) >> 16);
}
// order-preserving float<->uint encoding for atomicMax-based segment max
__device__ __forceinline__ unsigned encf(float f) {
  unsigned u = __float_as_uint(f);
  return (u & 0x80000000u) ? ~u : (u | 0x80000000u);
}
__device__ __forceinline__ float decf(unsigned u) {
  return __uint_as_float((u & 0x80000000u) ? (u & 0x7FFFFFFFu) : ~u);
}

// B fragment (32x16 bf16), prepacked in register order: 8 dwords/lane, lane-contiguous
__device__ __forceinline__ v16bf loadB(const uint4* wf, int fragIdx, int lane) {
  FragU f;
  const uint4* p = wf + ((size_t)fragIdx * 32 + lane) * 2;
  f.q[0] = p[0];
  f.q[1] = p[1];
  return f.v;
}
// A fragment (16x32 bf16) from row-major LDS H[rows][128]:
// VGPRs 0-3 = K k0..k0+7, VGPRs 4-7 = K k0+16..k0+23  -> two 16B-aligned b128 loads
__device__ __forceinline__ v16bf loadA_lds(const unsigned short* H, int rowIdx, int half, int ks) {
  FragU f;
  const uint4* p = (const uint4*)(H + rowIdx * 128 + ks * 32 + half * 8);
  f.q[0] = p[0];
  f.q[1] = p[2];   // +16 halfs
  return f.v;
}

// ---------------- prep kernels ----------------

// pack W[K][Ncols] f32 -> bf16 B-fragment order
__global__ void pack_w(const float* __restrict__ W, unsigned* __restrict__ dst, int K, int Ncols) {
  int tid = blockIdx.x * blockDim.x + threadIdx.x;
  int total = (Ncols / 16) * (K / 32) * 256;
  if (tid >= total) return;
  int v = tid & 7, lane = (tid >> 3) & 31, f = tid >> 8;
  int KS = K / 32, nc = f / KS, ks = f % KS;
  int col = nc * 16 + (lane & 15);
  int k0 = ks * 32 + (v < 4 ? 2 * v : 16 + 2 * (v - 4)) + (lane >> 4) * 8;
  dst[tid] = pk_bf16(W[k0 * Ncols + col], W[(k0 + 1) * Ncols + col]);
}

// pack edge_data[E][32] f32 -> bf16 A-fragment order per 16-edge tile
__global__ void pack_x(const float* __restrict__ X, unsigned* __restrict__ dst, int E) {
  int tid = blockIdx.x * blockDim.x + threadIdx.x;
  int total = (E / 16) * 256;
  if (tid >= total) return;
  int v = tid & 7, lane = (tid >> 3) & 31, t = tid >> 8;
  int row = lane & 15, half = lane >> 4;
  int e = t * 16 + row;
  int k0 = (v < 4 ? 2 * v : 16 + 2 * (v - 4)) + half * 8;
  dst[tid] = pk_bf16(X[e * 32 + k0], X[e * 32 + k0 + 1]);
}

__global__ void zero_kernel(float* p, int n) {
  for (int i = blockIdx.x * blockDim.x + threadIdx.x; i < n; i += gridDim.x * blockDim.x) p[i] = 0.0f;
}

// t = node @ W_input / 8   [N,16]
__global__ void calc_t(const float* __restrict__ node, const float* __restrict__ Wi,
                       float* __restrict__ t, int N) {
  int tid = blockIdx.x * blockDim.x + threadIdx.x;
  if (tid >= N * 16) return;
  int n = tid >> 4, a = tid & 15;
  float acc = 0.f;
#pragma unroll 8
  for (int c = 0; c < 64; ++c) acc += node[n * 64 + c] * Wi[c * 16 + a];
  t[tid] = acc * 0.125f;
}

// p = (t @ W_query / 4) @ W_dot   [N,8]
__global__ void calc_p(const float* __restrict__ t, const float* __restrict__ Wq,
                       const float* __restrict__ Wd, float* __restrict__ p, int N) {
  int tid = blockIdx.x * blockDim.x + threadIdx.x;
  if (tid >= N * 8) return;
  int n = tid >> 3, b = tid & 7;
  float acc = 0.f;
#pragma unroll
  for (int i = 0; i < 8; ++i) {
    float qi = 0.f;
#pragma unroll
    for (int a = 0; a < 16; ++a) qi += t[n * 16 + a] * Wq[a * 8 + i];
    acc += qi * 0.25f * Wd[i * 8 + b];
  }
  p[tid] = acc;
}

// ---------------- fused per-edge WMMA kernel ----------------
// One wave (32 threads) per block handles 64 edges (4 sub-tiles of 16):
// every B fragment fetched once from L2 and reused 4x.
// nc loops capped at unroll 2: enough to overlap B loads with WMMA while
// keeping the kernel <256 VGPRs and I-cache resident.
__global__ __launch_bounds__(32) void edge_kernel(
    const int* __restrict__ eidx, const float* __restrict__ t,
    const float* __restrict__ esh, const uint4* __restrict__ xfrag,
    const uint4* __restrict__ w1k, const float* __restrict__ b1k,
    const uint4* __restrict__ w2k, const float* __restrict__ b2k,
    const uint4* __restrict__ w3k, const float* __restrict__ b3k,
    const uint4* __restrict__ w1v, const float* __restrict__ b1v,
    const uint4* __restrict__ w2v, const float* __restrict__ b2v,
    const uint4* __restrict__ w3v, const float* __restrict__ b3v,
    float* __restrict__ kvec, float* __restrict__ vbuf, int E) {
  __shared__ unsigned short H1s[64 * 128];  // 16 KB
  __shared__ unsigned short H2s[64 * 128];  // 16 KB
  __shared__ float o_t[64][64];             // outer(t[src], sh)/8, [oidx][edge]  16 KB

  const int lane = threadIdx.x;
  const int half = lane >> 4, row = lane & 15;
  const int w = blockIdx.x;
  const int e0 = w * 64;

  // build o_t once: o_t[a*4+s][edge] = t[src][a]*sh[s]/8   (scale 1/sqrt(C_TP*C_SH))
  for (int ee = lane; ee < 64; ee += 32) {
    int e = e0 + ee;
    const float4 sh4 = ((const float4*)esh)[e];
    const int se = eidx[e];
    const float4* tp = (const float4*)(t + se * 16);
#pragma unroll
    for (int aq = 0; aq < 4; ++aq) {
      float4 tv = tp[aq];
      float ta[4] = {tv.x, tv.y, tv.z, tv.w};
#pragma unroll
      for (int ai = 0; ai < 4; ++ai) {
        int a = aq * 4 + ai;
        float v = ta[ai] * 0.125f;
        o_t[a * 4 + 0][ee] = v * sh4.x;
        o_t[a * 4 + 1][ee] = v * sh4.y;
        o_t[a * 4 + 2][ee] = v * sh4.z;
        o_t[a * 4 + 3][ee] = v * sh4.w;
      }
    }
  }
  // edge-feature A fragments (prepacked bf16), one per sub-tile
  v16bf A1[ST];
#pragma unroll
  for (int st = 0; st < ST; ++st) A1[st] = loadB(xfrag, w * ST + st, lane);
  __syncthreads();

  // shared 2-layer MLP front (layers 1&2), B reused across the 4 sub-tiles
  auto run_front = [&](const uint4* w1f, const float* b1, const uint4* w2f, const float* b2) {
#pragma unroll 2
    for (int nc = 0; nc < 8; ++nc) {          // layer 1: 32 -> 128
      v16bf B = loadB(w1f, nc, lane);
      float bb = b1[nc * 16 + row];
#pragma unroll
      for (int st = 0; st < ST; ++st) {
        v8f c = {};
        c = wmma_bf16(A1[st], B, c);
#pragma unroll
        for (int r = 0; r < 8; ++r)
          H1s[(st * 16 + half * 8 + r) * 128 + nc * 16 + row] = bf16s(fmaxf(c[r] + bb, 0.f));
      }
    }
    __syncthreads();
#pragma unroll 2
    for (int nc = 0; nc < 8; ++nc) {          // layer 2: 128 -> 128
      v16bf B0 = loadB(w2f, nc * 4 + 0, lane);
      v16bf B1 = loadB(w2f, nc * 4 + 1, lane);
      v16bf B2 = loadB(w2f, nc * 4 + 2, lane);
      v16bf B3 = loadB(w2f, nc * 4 + 3, lane);
      float bb = b2[nc * 16 + row];
#pragma unroll
      for (int st = 0; st < ST; ++st) {
        int rb = st * 16 + row;
        v8f c = {};
        c = wmma_bf16(loadA_lds(H1s, rb, half, 0), B0, c);
        c = wmma_bf16(loadA_lds(H1s, rb, half, 1), B1, c);
        c = wmma_bf16(loadA_lds(H1s, rb, half, 2), B2, c);
        c = wmma_bf16(loadA_lds(H1s, rb, half, 3), B3, c);
#pragma unroll
        for (int r = 0; r < 8; ++r)
          H2s[(st * 16 + half * 8 + r) * 128 + nc * 16 + row] = bf16s(fmaxf(c[r] + bb, 0.f));
      }
    }
    __syncthreads();
  };

  // ---------------- key path: layer3 128->512, fused bilinear contraction ----------------
  run_front(w1k, b1k, w2k, b2k);
  {
    float kacc[ST][8] = {};
#pragma unroll 2
    for (int nc = 0; nc < 32; ++nc) {
      v16bf B0 = loadB(w3k, nc * 4 + 0, lane);
      v16bf B1 = loadB(w3k, nc * 4 + 1, lane);
      v16bf B2 = loadB(w3k, nc * 4 + 2, lane);
      v16bf B3 = loadB(w3k, nc * 4 + 3, lane);
      float bb = b3k[nc * 16 + row];
      int oi = nc * 2 + (row >> 3);           // o-index = j>>3, j = nc*16+row
#pragma unroll
      for (int st = 0; st < ST; ++st) {
        int rb = st * 16 + row;
        v8f c = {};
        c = wmma_bf16(loadA_lds(H2s, rb, half, 0), B0, c);
        c = wmma_bf16(loadA_lds(H2s, rb, half, 1), B1, c);
        c = wmma_bf16(loadA_lds(H2s, rb, half, 2), B2, c);
        c = wmma_bf16(loadA_lds(H2s, rb, half, 3), B3, c);
        const float* orow = &o_t[oi][st * 16 + half * 8];
#pragma unroll
        for (int r = 0; r < 8; ++r) {
          float pv = (c[r] + bb) * orow[r];
          pv += __shfl_xor(pv, 8, 32);        // pair of columns sharing b = row&7
          kacc[st][r] += pv;
        }
      }
    }
    if (row < 8) {
#pragma unroll
      for (int st = 0; st < ST; ++st)
#pragma unroll
        for (int r = 0; r < 8; ++r)
          kvec[(size_t)(e0 + st * 16 + half * 8 + r) * 8 + row] = kacc[st][r];
    }
  }
  __syncthreads();

  // ---------------- value path: layer3 128->1024, fused contraction ----------------
  run_front(w1v, b1v, w2v, b2v);
  {
    float vacc[ST][8] = {};
#pragma unroll 2
    for (int nc = 0; nc < 64; ++nc) {
      v16bf B0 = loadB(w3v, nc * 4 + 0, lane);
      v16bf B1 = loadB(w3v, nc * 4 + 1, lane);
      v16bf B2 = loadB(w3v, nc * 4 + 2, lane);
      v16bf B3 = loadB(w3v, nc * 4 + 3, lane);
      float bb = b3v[nc * 16 + row];
#pragma unroll
      for (int st = 0; st < ST; ++st) {
        int rb = st * 16 + row;
        v8f c = {};
        c = wmma_bf16(loadA_lds(H2s, rb, half, 0), B0, c);
        c = wmma_bf16(loadA_lds(H2s, rb, half, 1), B1, c);
        c = wmma_bf16(loadA_lds(H2s, rb, half, 2), B2, c);
        c = wmma_bf16(loadA_lds(H2s, rb, half, 3), B3, c);
        const float* orow = &o_t[nc][st * 16 + half * 8]; // o-index = j>>4 = nc; b = row
#pragma unroll
        for (int r = 0; r < 8; ++r) vacc[st][r] += (c[r] + bb) * orow[r];
      }
    }
#pragma unroll
    for (int st = 0; st < ST; ++st)
#pragma unroll
      for (int r = 0; r < 8; ++r)
        vbuf[(size_t)(e0 + st * 16 + half * 8 + r) * 16 + row] = vacc[st][r];
  }
}

// ---------------- attention logit + segment max ----------------
__global__ void logit_kernel(const int* __restrict__ eidx, const float* __restrict__ pfeat,
                             const float* __restrict__ kvec, float* __restrict__ abuf,
                             unsigned* __restrict__ menc, int E) {
  int e = blockIdx.x * blockDim.x + threadIdx.x;
  if (e >= E) return;
  int d = eidx[E + e];
  float acc = 0.f;
#pragma unroll
  for (int b = 0; b < 8; ++b) acc += pfeat[d * 8 + b] * kvec[(size_t)e * 8 + b];
  acc *= 0.125f;   // 1/sqrt(C_KEY*C_KEY)
  abuf[e] = acc;
  atomicMax(&menc[d], encf(acc));
}

// ---------------- softmax + aggregation ----------------
__global__ void softmax_agg(const int* __restrict__ eidx, const float* __restrict__ abuf,
                            const float* __restrict__ vbuf, const unsigned* __restrict__ menc,
                            float* __restrict__ z, float* __restrict__ agg, int E) {
  int tid = blockIdx.x * blockDim.x + threadIdx.x;
  int c = tid & 15, e = tid >> 4;
  if (e >= E) return;
  int d = eidx[E + e];
  float w = expf(abuf[e] - decf(menc[d]));
  if (c == 0) unsafeAtomicAdd(&z[d], w);
  unsafeAtomicAdd(&agg[d * 16 + c], w * vbuf[(size_t)e * 16 + c]);
}

// out_pre = (agg/z) @ W_output / 4 + residual; accumulate BN stats
__global__ __launch_bounds__(64) void node_out(const float* __restrict__ agg, const float* __restrict__ z,
                                               const float* __restrict__ Wo, const float* __restrict__ node,
                                               float* __restrict__ pre, float* __restrict__ sums,
                                               float* __restrict__ sumsq, int N) {
  int n = blockIdx.x, c = threadIdx.x;
  float zz = z[n];
  float zi = zz > 0.f ? 1.0f / zz : 0.0f;
  float acc = 0.f;
#pragma unroll
  for (int j = 0; j < 16; ++j) acc += agg[n * 16 + j] * Wo[j * 64 + c];
  float x = acc * zi * 0.25f + node[n * 64 + c];
  pre[n * 64 + c] = x;
  unsafeAtomicAdd(&sums[c], x);
  unsafeAtomicAdd(&sumsq[c], x * x);
}

__global__ void bn_stats(const float* sums, const float* sumsq, const float* bw, const float* bb,
                         float* scale, float* shift, int N) {
  int c = threadIdx.x;
  float mean = sums[c] / (float)N;
  float var = sumsq[c] / (float)N - mean * mean;
  float sc = rsqrtf(var + 1e-5f) * bw[c];
  scale[c] = sc;
  shift[c] = bb[c] - mean * sc;
}

__global__ void bn_norm(const float* __restrict__ pre, const float* __restrict__ scale,
                        const float* __restrict__ shift, float* __restrict__ out, int total) {
  int i = blockIdx.x * blockDim.x + threadIdx.x;
  if (i >= total) return;
  int c = i & 63;
  out[i] = pre[i] * scale[c] + shift[c];
}

// ---------------- host launcher ----------------
extern "C" void kernel_launch(void* const* d_in, const int* in_sizes, int n_in,
                              void* d_out, int out_size, void* d_ws, size_t ws_size,
                              hipStream_t stream) {
  const float* node = (const float*)d_in[0];
  const int*   eidx = (const int*)d_in[1];
  const float* edata = (const float*)d_in[2];
  const float* esh = (const float*)d_in[3];
  const float* Wi = (const float*)d_in[4];
  const float* Wq = (const float*)d_in[5];
  const float* Wd = (const float*)d_in[6];
  const float* Wo = (const float*)d_in[7];
  const float* Wk1 = (const float*)d_in[8];  const float* bk1 = (const float*)d_in[9];
  const float* Wk2 = (const float*)d_in[10]; const float* bk2 = (const float*)d_in[11];
  const float* Wk3 = (const float*)d_in[12]; const float* bk3 = (const float*)d_in[13];
  const float* Wv1 = (const float*)d_in[14]; const float* bv1 = (const float*)d_in[15];
  const float* Wv2 = (const float*)d_in[16]; const float* bv2 = (const float*)d_in[17];
  const float* Wv3 = (const float*)d_in[18]; const float* bv3 = (const float*)d_in[19];
  const float* bnw = (const float*)d_in[20]; const float* bnb = (const float*)d_in[21];
  float* out = (float*)d_out;

  char* ws = (char*)d_ws;
  size_t off = 0;
  auto alloc = [&](size_t bytes) -> char* {
    char* p = ws + off;
    off += (bytes + 255) & ~(size_t)255;
    return p;
  };
  float*    tb   = (float*)alloc((size_t)NN * 16 * 4);
  float*    pb   = (float*)alloc((size_t)NN * 8 * 4);
  unsigned* xf   = (unsigned*)alloc((size_t)EE * 16 * 4);
  unsigned* f1k  = (unsigned*)alloc(2048 * 4);
  unsigned* f2k  = (unsigned*)alloc(8192 * 4);
  unsigned* f3k  = (unsigned*)alloc(32768 * 4);
  unsigned* f1v  = (unsigned*)alloc(2048 * 4);
  unsigned* f2v  = (unsigned*)alloc(8192 * 4);
  unsigned* f3v  = (unsigned*)alloc(65536 * 4);
  float*    kvec = (float*)alloc((size_t)EE * 8 * 4);
  float*    abuf = (float*)alloc((size_t)EE * 4);
  float*    vbuf = (float*)alloc((size_t)EE * 16 * 4);
  float*    pre  = (float*)alloc((size_t)NN * 64 * 4);
  float*    scl  = (float*)alloc(64 * 4);
  float*    shf  = (float*)alloc(64 * 4);
  char*     zg   = alloc((size_t)(NN + NN + NN * 16 + 128) * 4);
  unsigned* menc = (unsigned*)zg;
  float*    zb   = (float*)(menc + NN);
  float*    agg  = zb + NN;
  float*    sums = agg + NN * 16;
  float*    sumsq = sums + 64;
  int zn = NN * 18 + 128;

  zero_kernel<<<256, 256, 0, stream>>>((float*)menc, zn);
  pack_w<<<8, 256, 0, stream>>>(Wk1, f1k, 32, 128);
  pack_w<<<32, 256, 0, stream>>>(Wk2, f2k, 128, 128);
  pack_w<<<128, 256, 0, stream>>>(Wk3, f3k, 128, 512);
  pack_w<<<8, 256, 0, stream>>>(Wv1, f1v, 32, 128);
  pack_w<<<32, 256, 0, stream>>>(Wv2, f2v, 128, 128);
  pack_w<<<256, 256, 0, stream>>>(Wv3, f3v, 128, 1024);
  pack_x<<<(EE * 16) / 256, 256, 0, stream>>>(edata, xf, EE);
  calc_t<<<(NN * 16 + 255) / 256, 256, 0, stream>>>(node, Wi, tb, NN);
  calc_p<<<(NN * 8 + 255) / 256, 256, 0, stream>>>(tb, Wq, Wd, pb, NN);

  edge_kernel<<<EE / 64, 32, 0, stream>>>(
      eidx, tb, esh, (const uint4*)xf,
      (const uint4*)f1k, bk1, (const uint4*)f2k, bk2, (const uint4*)f3k, bk3,
      (const uint4*)f1v, bv1, (const uint4*)f2v, bv2, (const uint4*)f3v, bv3,
      kvec, vbuf, EE);

  logit_kernel<<<(EE + 255) / 256, 256, 0, stream>>>(eidx, pb, kvec, abuf, menc, EE);
  softmax_agg<<<(EE * 16) / 256, 256, 0, stream>>>(eidx, abuf, vbuf, menc, zb, agg, EE);
  node_out<<<NN, 64, 0, stream>>>(agg, zb, Wo, node, pre, sums, sumsq, NN);
  bn_stats<<<1, 64, 0, stream>>>(sums, sumsq, bnw, bnb, scl, shf, NN);
  bn_norm<<<(NN * 64 + 255) / 256, 256, 0, stream>>>(pre, scl, shf, out, NN * 64);
}
